// Convolution_26147760898303
// MI455X (gfx1250) — matrix operand order repaired and checked by reading, not compiled, not run on hardware
//
#include <hip/hip_runtime.h>

typedef __attribute__((ext_vector_type(8)))  __bf16 v8bf;
typedef __attribute__((ext_vector_type(16))) __bf16 v16bf;
typedef __attribute__((ext_vector_type(8)))  float  v8f;

#define C_IN   128
#define K_OUT  256
#define HW_DIM 56
#define IMG    (HW_DIM * HW_DIM)   // 3136
#define KRED   (C_IN * 9)          // 1152 ; permuted order t = rs*128 + c
#define BM     128                 // out-channels per block
#define BN     64                  // pixels per block
#define KC     32                  // reduction chunk == WMMA K
#define LDAB   40                  // padded bf16 row stride (16B aligned, bank-safe)
#define LDBB   40

// Truncation-based fp32 -> (bf16 hi, bf16 lo) split; hi+lo keeps ~16 mantissa bits.
__device__ __forceinline__ void split1(float x, unsigned short& hi, unsigned short& lo) {
    unsigned u = __builtin_bit_cast(unsigned, x);
    unsigned h = u & 0xFFFF0000u;
    hi = (unsigned short)(h >> 16);
    float r = x - __builtin_bit_cast(float, h);
    lo = (unsigned short)(__builtin_bit_cast(unsigned, r) >> 16);
}

// Split two floats -> packed hi-pair / lo-pair (elem0 in [15:0]).
__device__ __forceinline__ void split2(float x0, float x1, unsigned& hi, unsigned& lo) {
    unsigned u0 = __builtin_bit_cast(unsigned, x0);
    unsigned u1 = __builtin_bit_cast(unsigned, x1);
    unsigned h0 = u0 & 0xFFFF0000u;
    unsigned h1 = u1 & 0xFFFF0000u;
    hi = h1 | (h0 >> 16);
    float r0 = x0 - __builtin_bit_cast(float, h0);
    float r1 = x1 - __builtin_bit_cast(float, h1);
    lo = (__builtin_bit_cast(unsigned, r1) & 0xFFFF0000u) |
         (__builtin_bit_cast(unsigned, r0) >> 16);
}

__device__ __forceinline__ v16bf ld_frag2(const unsigned short* p0, const unsigned short* p1) {
    v8bf a = *(const v8bf*)p0;
    v8bf b = *(const v8bf*)p1;
    return __builtin_shufflevector(a, b, 0, 1, 2, 3, 4, 5, 6, 7,
                                         8, 9, 10, 11, 12, 13, 14, 15);
}

// -------- pre-pass: split weights to bf16 hi/lo in permuted layout [k][rs*128+c] --------
__global__ __launch_bounds__(256) void split_weights(
    const float* __restrict__ Wt,        // [256][1152] as OIHW (t_src = c*9+rs)
    unsigned short* __restrict__ Whi,    // [256][1152] permuted (t_dst = rs*128+c)
    unsigned short* __restrict__ Wlo)
{
    int e  = blockIdx.x * 256 + threadIdx.x;   // 0 .. 256*1152-1
    int k  = e / KRED;
    int td = e - k * KRED;
    int rs = td >> 7;
    int c  = td & 127;
    float v = Wt[(long)k * KRED + c * 9 + rs];
    unsigned short h, l;
    split1(v, h, l);
    Whi[e] = h;
    Wlo[e] = l;
}

// -------- main conv: implicit GEMM, bf16x3 WMMA, async-LDS weight staging --------
__global__ __launch_bounds__(256) void conv3x3_wmma_bf16x3(
    const float* __restrict__ X,            // [32,128,56,56]
    const unsigned short* __restrict__ Whi, // [256][1152] bf16 hi (permuted)
    const unsigned short* __restrict__ Wlo, // [256][1152] bf16 lo
    const float* __restrict__ bias,         // [256]
    float* __restrict__ out)                // [32,256,56,56]
{
    __shared__ unsigned short AsH[BM * LDAB];  // weights hi  (128 x 40 bf16)
    __shared__ unsigned short AsL[BM * LDAB];  // weights lo
    __shared__ unsigned short BsH[BN * LDBB];  // im2col  hi  [pixel][t]
    __shared__ unsigned short BsL[BN * LDBB];  // im2col  lo

    const int tid  = threadIdx.x;
    const int wv   = tid >> 5;
    const int lane = tid & 31;
    const int half = lane >> 4;
    const int lm   = lane & 15;

    const int p0 = blockIdx.x * BN;
    const int m0 = blockIdx.y * BM;

    // ---- per-thread fixed geometry: one pixel, 8 consecutive t values ----
    const int pl = tid & 63;              // pixel-local column
    const int tb = (tid >> 6) << 3;       // t-local base: 0,8,16,24
    const int p  = p0 + pl;
    const int pn = p / IMG;
    const int phw = p - pn * IMG;
    const int ph = phw / HW_DIM;
    const int pw = phw - ph * HW_DIM;
    const long xbase = (long)pn * C_IN * IMG;

    // A staging via async-LDS: this thread's two 16B chunks per matrix
    const int arow0 = (tid + 0)   >> 2;           // idx>>2, idx=tid, tid+256
    const int acol0 = ((tid + 0)  & 3) << 3;
    const int arow1 = (tid + 256) >> 2;
    const int acol1 = ((tid + 256) & 3) << 3;

    v8f acc0 = {}, acc1 = {}, acc2 = {}, acc3 = {};

    for (int t0 = 0; t0 < KRED; t0 += KC) {
        // tap (r,s) is uniform across the whole chunk in permuted order
        const int rs = t0 >> 7;           // 0..8
        const int c0 = t0 & 127;          // channel base of chunk
        const int r  = (rs * 11) >> 5;    // rs/3 for rs<9
        const int s  = rs - 3 * r;

        // ---- stage A (async): pre-split bf16 weights -> LDS, no VGPR round-trip ----
        {
            unsigned ldsA0 = (unsigned)(size_t)&AsH[arow0 * LDAB + acol0];
            unsigned long gA0 = (unsigned long)(size_t)(Whi + (long)(m0 + arow0) * KRED + t0 + acol0);
            asm volatile("global_load_async_to_lds_b128 %0, %1, off" :: "v"(ldsA0), "v"(gA0) : "memory");
            unsigned ldsA1 = (unsigned)(size_t)&AsH[arow1 * LDAB + acol1];
            unsigned long gA1 = (unsigned long)(size_t)(Whi + (long)(m0 + arow1) * KRED + t0 + acol1);
            asm volatile("global_load_async_to_lds_b128 %0, %1, off" :: "v"(ldsA1), "v"(gA1) : "memory");
            unsigned ldsB0 = (unsigned)(size_t)&AsL[arow0 * LDAB + acol0];
            unsigned long gB0 = (unsigned long)(size_t)(Wlo + (long)(m0 + arow0) * KRED + t0 + acol0);
            asm volatile("global_load_async_to_lds_b128 %0, %1, off" :: "v"(ldsB0), "v"(gB0) : "memory");
            unsigned ldsB1 = (unsigned)(size_t)&AsL[arow1 * LDAB + acol1];
            unsigned long gB1 = (unsigned long)(size_t)(Wlo + (long)(m0 + arow1) * KRED + t0 + acol1);
            asm volatile("global_load_async_to_lds_b128 %0, %1, off" :: "v"(ldsB1), "v"(gB1) : "memory");
        }

        // ---- stage B: 8 elems (one pixel, t=tb..tb+7), one bounds check ----
        {
            const int ih = ph + r - 1;
            const int iw = pw + s - 1;
            const bool inb = ((unsigned)ih < HW_DIM) & ((unsigned)iw < HW_DIM);
            const float* Xp = X + xbase + (long)(c0 + tb) * IMG + ih * HW_DIM + iw;
            float xv[8];
            #pragma unroll
            for (int i = 0; i < 8; ++i)
                xv[i] = inb ? Xp[(long)i * IMG] : 0.0f;
            unsigned h01, l01, h23, l23, h45, l45, h67, l67;
            split2(xv[0], xv[1], h01, l01);
            split2(xv[2], xv[3], h23, l23);
            split2(xv[4], xv[5], h45, l45);
            split2(xv[6], xv[7], h67, l67);
            uint4 hv = {h01, h23, h45, h67};
            uint4 lv = {l01, l23, l45, l67};
            *(uint4*)&BsH[pl * LDBB + tb] = hv;
            *(uint4*)&BsL[pl * LDBB + tb] = lv;
        }

        asm volatile("s_wait_asynccnt 0" ::: "memory");
        __syncthreads();

        // ---- A frags (16x32 bf16: lane<16 K=0-7,16-23; lane>=16 K=8-15,24-31) ----
        const int arow = (wv * 16 + lm) * LDAB;
        const int lowK = half * 8;
        v16bf ahi = ld_frag2(&AsH[arow + lowK], &AsH[arow + lowK + 16]);
        v16bf alo = ld_frag2(&AsL[arow + lowK], &AsL[arow + lowK + 16]);

        // ---- 4 N-tiles x 3 WMMAs (hi*hi + hi*lo + lo*hi), f32 accumulate ----
        #pragma unroll
        for (int j = 0; j < 4; ++j) {
            const int brow = (j * 16 + lm) * LDBB + half * 8;  // lane<16: K=0-7,16-23; else 8-15,24-31
            v16bf bhi = ld_frag2(&BsH[brow], &BsH[brow + 16]);
            v16bf blo = ld_frag2(&BsL[brow], &BsL[brow + 16]);
            v8f* accp = (j == 0) ? &acc0 : (j == 1) ? &acc1 : (j == 2) ? &acc2 : &acc3;
            v8f a = *accp;
            a = __builtin_amdgcn_wmma_f32_16x16x32_bf16(false, ahi, false, bhi, (short)0, a, false, false);
            a = __builtin_amdgcn_wmma_f32_16x16x32_bf16(false, ahi, false, blo, (short)0, a, false, false);
            a = __builtin_amdgcn_wmma_f32_16x16x32_bf16(false, alo, false, bhi, (short)0, a, false, false);
            *accp = a;
        }
        __syncthreads();
    }

    // ---- epilogue: C/D layout (VGPR r: lane<16 -> M=r, lane>=16 -> M=r+8) + bias ----
    #pragma unroll
    for (int rr = 0; rr < 8; ++rr) {
        int M   = wv * 16 + rr + half * 8;
        int kch = m0 + M;
        float bv = bias[kch];
        #pragma unroll
        for (int j = 0; j < 4; ++j) {
            const v8f& a = (j == 0) ? acc0 : (j == 1) ? acc1 : (j == 2) ? acc2 : acc3;
            int pp = p0 + j * 16 + lm;
            int n  = pp / IMG;
            int hw = pp - n * IMG;
            out[((long)n * K_OUT + kch) * IMG + hw] = a[rr] + bv;
        }
    }
}

extern "C" void kernel_launch(void* const* d_in, const int* in_sizes, int n_in,
                              void* d_out, int out_size, void* d_ws, size_t ws_size,
                              hipStream_t stream) {
    const float* x  = (const float*)d_in[0];
    const float* w  = (const float*)d_in[1];
    const float* b  = (const float*)d_in[2];
    float* out = (float*)d_out;

    // workspace: Whi + Wlo, 2 * 256*1152 bf16 = 1.18 MB (written every call)
    unsigned short* Whi = (unsigned short*)d_ws;
    unsigned short* Wlo = Whi + (size_t)K_OUT * KRED;

    split_weights<<<dim3(K_OUT * KRED / 256), dim3(256), 0, stream>>>(w, Whi, Wlo);

    // total pixels = 32*3136 = 100352 ; grid = (100352/64, 256/128)
    dim3 grid(100352 / BN, K_OUT / BM, 1);
    dim3 block(256, 1, 1);
    conv3x3_wmma_bf16x3<<<grid, block, 0, stream>>>(x, Whi, Wlo, b, out);
}
